// RowInteraction_10247791968423
// MI455X (gfx1250) — matrix operand order repaired and checked by reading, not compiled, not run on hardware
//
#include <hip/hip_runtime.h>

typedef _Float16 v16h __attribute__((ext_vector_type(16)));
typedef float    v8f  __attribute__((ext_vector_type(8)));

#define SEQ   64
#define EDIM  128
#define NHEAD 4
#define DHEAD 32
#define FFDIM 256
#define NLAYER 3
#define CTOK  4

// packed-weight workspace offsets (dwords)
#define PK_EE   (NLAYER * 8 * 4 * 256)    // E x E matrix: 8 ntiles, 4 kchunks
#define OFF_Q   0
#define OFF_K   (PK_EE)
#define OFF_V   (2 * PK_EE)
#define OFF_O   (3 * PK_EE)
#define OFF_W1  (4 * PK_EE)                       // 128x256: 16 nt, 4 kc
#define PK_W1   (NLAYER * 16 * 4 * 256)
#define OFF_W2  (OFF_W1 + PK_W1)                  // 256x128: 8 nt, 8 kc
#define PK_W2   (NLAYER * 8 * 8 * 256)

union FragU { v16h v; unsigned u[8]; };
union H2U  { _Float16 h[2]; unsigned u; };

__device__ __forceinline__ v8f wmma_f16(v16h a, v16h b, v8f c) {
  return __builtin_amdgcn_wmma_f32_16x16x32_f16(false, a, false, b, (short)0, c,
                                                false, false);
}

// ============ weight pre-pack: f32 row-major -> f16 B-fragment layout ========
__global__ __launch_bounds__(256) void prepack_kernel(const float* __restrict__ W,
                                                      unsigned* __restrict__ dst,
                                                      int K, int N) {
  int id   = blockIdx.x * 256 + threadIdx.x;
  int i    = id & 7;
  int lane = (id >> 3) & 31;
  int frag = id >> 8;
  int K32  = K >> 5, N16 = N >> 4;
  int kc   = frag % K32;
  int nt   = (frag / K32) % N16;
  int l    = frag / (K32 * N16);
  int n    = nt * 16 + (lane & 15);
  int k0   = kc * 32 + (lane >> 4) * 16 + 2 * i;
  const float* src = W + (size_t)l * K * N;
  H2U p;
  p.h[0] = (_Float16)src[(size_t)k0 * N + n];
  p.h[1] = (_Float16)src[(size_t)(k0 + 1) * N + n];
  dst[id] = p.u;
}

// ============ fragment loaders ==============================================
__device__ __forceinline__ v16h load_a(const _Float16* A, int ald, int mb, int kb0) {
  int lane = threadIdx.x & 31;
  int m  = mb + (lane & 15);
  int hl = lane >> 4;
  const uint4* rp = (const uint4*)(A + m * ald + kb0 + hl * 16);
  FragU f;
  uint4 lo = rp[0];   // K = kb0 + hl*16 .. +7
  uint4 hi = rp[1];   // K = kb0 + hl*16 + 8 .. +15
  f.u[0] = lo.x; f.u[1] = lo.y; f.u[2] = lo.z; f.u[3] = lo.w;
  f.u[4] = hi.x; f.u[5] = hi.y; f.u[6] = hi.z; f.u[7] = hi.w;
  return f.v;
}

__device__ __forceinline__ v16h load_b_pk(const unsigned* __restrict__ P, int frag) {
  int lane = threadIdx.x & 31;
  const uint4* p = (const uint4*)(P + frag * 256 + lane * 8);
  FragU f;
  uint4 a = p[0], b = p[1];
  f.u[0] = a.x; f.u[1] = a.y; f.u[2] = a.z; f.u[3] = a.w;
  f.u[4] = b.x; f.u[5] = b.y; f.u[6] = b.z; f.u[7] = b.w;
  return f.v;
}

__device__ __forceinline__ v16h load_b_kT(const _Float16* kbuf, int tb, int dbase) {
  int lane = threadIdx.x & 31;
  int t  = tb + (lane & 15);
  int hl = lane >> 4;
  const uint4* rp = (const uint4*)(kbuf + t * EDIM + dbase + hl * 16);
  FragU f;
  uint4 a = rp[0], b = rp[1];
  f.u[0] = a.x; f.u[1] = a.y; f.u[2] = a.z; f.u[3] = a.w;
  f.u[4] = b.x; f.u[5] = b.y; f.u[6] = b.z; f.u[7] = b.w;
  return f.v;
}

__device__ __forceinline__ v16h load_b_vT(const _Float16* vt, int kb0, int nb) {
  int lane = threadIdx.x & 31;
  int n  = nb + (lane & 15);
  int hl = lane >> 4;
  const uint4* rp = (const uint4*)(vt + n * SEQ + kb0 + hl * 16);
  FragU f;
  uint4 a = rp[0], b = rp[1];
  f.u[0] = a.x; f.u[1] = a.y; f.u[2] = a.z; f.u[3] = a.w;
  f.u[4] = b.x; f.u[5] = b.y; f.u[6] = b.z; f.u[7] = b.w;
  return f.v;
}

__device__ __forceinline__ void store_h(v8f c, _Float16* out, int old, int mb, int nb) {
  int lane = threadIdx.x & 31;
  int n  = nb + (lane & 15);
  int m0 = mb + 8 * (lane >> 4);
#pragma unroll
  for (int v = 0; v < 8; ++v) out[(m0 + v) * old + n] = (_Float16)c[v];
}

__device__ __forceinline__ void store_h_T(v8f c, _Float16* vt, int mb, int nb) {
  int lane = threadIdx.x & 31;
  int n  = nb + (lane & 15);
  int m0 = mb + 8 * (lane >> 4);
  H2U p0, p1, p2, p3;
  p0.h[0] = (_Float16)c[0]; p0.h[1] = (_Float16)c[1];
  p1.h[0] = (_Float16)c[2]; p1.h[1] = (_Float16)c[3];
  p2.h[0] = (_Float16)c[4]; p2.h[1] = (_Float16)c[5];
  p3.h[0] = (_Float16)c[6]; p3.h[1] = (_Float16)c[7];
  uint4 q; q.x = p0.u; q.y = p1.u; q.z = p2.u; q.w = p3.u;
  *(uint4*)(vt + n * SEQ + m0) = q;
}

__device__ __forceinline__ void add_xs(v8f c, float* xs, int mb, int nb) {
  int lane = threadIdx.x & 31;
  int n  = nb + (lane & 15);
  int m0 = mb + 8 * (lane >> 4);
#pragma unroll
  for (int v = 0; v < 8; ++v) xs[(m0 + v) * EDIM + n] += c[v];
}

__device__ __forceinline__ void gemm128(const _Float16* A, int ald,
                                        const unsigned* __restrict__ P,
                                        int ntile, int K32, int kc0,
                                        float bias, v8f acc[4]) {
  v16h bf[4];
#pragma unroll
  for (int kc = 0; kc < 4; ++kc) bf[kc] = load_b_pk(P, ntile * K32 + kc0 + kc);
#pragma unroll
  for (int rt = 0; rt < 4; ++rt) {
    v8f c;
#pragma unroll
    for (int j = 0; j < 8; ++j) c[j] = bias;
#pragma unroll
    for (int kc = 0; kc < 4; ++kc)
      c = wmma_f16(load_a(A, ald, rt * 16, kc * 32), bf[kc], c);
    acc[rt] = c;
  }
}

// ---- fully-parallel LayerNorm: 4 lanes per row, shfl_xor reductions ----
__device__ __forceinline__ void layer_norm_par(const float* xs, const float* g,
                                               const float* b, _Float16* out, int tid) {
  int row = tid >> 2, sub = tid & 3;
  const float4* r4 = (const float4*)(xs + row * EDIM + sub * 32);
  float4 v[8];
  float s = 0.f, ss = 0.f;
#pragma unroll
  for (int i = 0; i < 8; ++i) {
    v[i] = r4[i];
    s  += v[i].x + v[i].y + v[i].z + v[i].w;
    ss += v[i].x * v[i].x + v[i].y * v[i].y + v[i].z * v[i].z + v[i].w * v[i].w;
  }
  s  += __shfl_xor(s, 1);  ss += __shfl_xor(ss, 1);
  s  += __shfl_xor(s, 2);  ss += __shfl_xor(ss, 2);
  float m = s * (1.0f / EDIM);
  float r = rsqrtf(ss * (1.0f / EDIM) - m * m + 1e-5f);
  const float4* g4 = (const float4*)(g + sub * 32);
  const float4* b4 = (const float4*)(b + sub * 32);
  uint4* o4 = (uint4*)(out + row * EDIM + sub * 32);
#pragma unroll
  for (int i = 0; i < 4; ++i) {
    float4 va = v[2 * i], vb = v[2 * i + 1];
    float4 ga = g4[2 * i], gb = g4[2 * i + 1];
    float4 ba = b4[2 * i], bb = b4[2 * i + 1];
    H2U p0, p1, p2, p3;
    p0.h[0] = (_Float16)((va.x - m) * r * ga.x + ba.x);
    p0.h[1] = (_Float16)((va.y - m) * r * ga.y + ba.y);
    p1.h[0] = (_Float16)((va.z - m) * r * ga.z + ba.z);
    p1.h[1] = (_Float16)((va.w - m) * r * ga.w + ba.w);
    p2.h[0] = (_Float16)((vb.x - m) * r * gb.x + bb.x);
    p2.h[1] = (_Float16)((vb.y - m) * r * gb.y + bb.y);
    p3.h[0] = (_Float16)((vb.z - m) * r * gb.z + bb.z);
    p3.h[1] = (_Float16)((vb.w - m) * r * gb.w + bb.w);
    uint4 q; q.x = p0.u; q.y = p1.u; q.z = p2.u; q.w = p3.u;
    o4[i] = q;
  }
}

__global__ __launch_bounds__(256) void vit_cls_kernel(
    const float* __restrict__ emb,  const float* __restrict__ cls,
    const float* __restrict__ ln1g, const float* __restrict__ ln1b,
    const unsigned* __restrict__ wpk,
    const float* __restrict__ bq,   const float* __restrict__ bk,
    const float* __restrict__ bv,   const float* __restrict__ bo,
    const float* __restrict__ ln2g, const float* __restrict__ ln2b,
    const float* __restrict__ b1,   const float* __restrict__ b2,
    const float* __restrict__ olng, const float* __restrict__ olnb,
    const int*   __restrict__ dvec, float* __restrict__ out, int TT) {
  __shared__ float    xs[SEQ * EDIM];   // residual stream, f32
  __shared__ _Float16 hf[SEQ * EDIM];   // ln out / attn-o (A operand)
  __shared__ _Float16 qb[SEQ * EDIM];   // q  (gelu buffer in FFN)
  __shared__ _Float16 kb[SEQ * EDIM];   // k
  __shared__ _Float16 vt[EDIM * SEQ];   // V transposed: vt[n][t]
  __shared__ float    sf[SEQ * SEQ];    // scores f32
  __shared__ _Float16 pb[SEQ * SEQ];    // softmax probs f16
  __shared__ float    rct[SEQ * 16];    // rope cos table
  __shared__ float    rst[SEQ * 16];    // rope sin table

  const int n    = blockIdx.x;
  const int tid  = threadIdx.x;
  const int wid  = tid >> 5;
  const int lane = tid & 31;
  const int dC   = dvec[n / TT] + CTOK;

  // ---- load x (vectorized) + rope tables ----
  const float4* xin4 = (const float4*)(emb + (size_t)n * (SEQ * EDIM));
  const float4* cls4 = (const float4*)cls;
  float4* xs4 = (float4*)xs;
  for (int i = tid; i < SEQ * EDIM / 4; i += 256)
    xs4[i] = (i < CTOK * EDIM / 4) ? cls4[i] : xin4[i];
  for (int i = tid; i < SEQ * 16; i += 256) {
    int r = i >> 4, j = i & 15;
    float ang = (float)r * __powf(100000.0f, -(float)j * (1.0f / 16.0f));
    rct[i] = __cosf(ang);
    rst[i] = __sinf(ang);
  }
  __syncthreads();

  const float scale = 0.17677669529663687f;  // 1/sqrt(32)

  for (int l = 0; l < NLAYER; ++l) {
    const unsigned* pq = wpk + OFF_Q  + l * (8 * 4 * 256);
    const unsigned* pk = wpk + OFF_K  + l * (8 * 4 * 256);
    const unsigned* pv = wpk + OFF_V  + l * (8 * 4 * 256);
    const unsigned* po = wpk + OFF_O  + l * (8 * 4 * 256);
    const unsigned* p1 = wpk + OFF_W1 + l * (16 * 4 * 256);
    const unsigned* p2 = wpk + OFF_W2 + l * (8 * 8 * 256);

    // ---- LN1 ----
    layer_norm_par(xs, ln1g + l * EDIM, ln1b + l * EDIM, hf, tid);
    __syncthreads();

    // ---- Q, K, V projections ----
    {
      int nb = wid * 16;
      int nl = nb + (lane & 15);
      v8f acc[4];
      gemm128(hf, EDIM, pq, wid, 4, 0, bq[l * EDIM + nl], acc);
#pragma unroll
      for (int rt = 0; rt < 4; ++rt) store_h(acc[rt], qb, EDIM, rt * 16, nb);
      gemm128(hf, EDIM, pk, wid, 4, 0, bk[l * EDIM + nl], acc);
#pragma unroll
      for (int rt = 0; rt < 4; ++rt) store_h(acc[rt], kb, EDIM, rt * 16, nb);
      gemm128(hf, EDIM, pv, wid, 4, 0, bv[l * EDIM + nl], acc);
#pragma unroll
      for (int rt = 0; rt < 4; ++rt) store_h_T(acc[rt], vt, rt * 16, nb);
    }
    __syncthreads();

    // ---- RoPE on q and k ----
    for (int i = tid; i < 2 * SEQ * 64; i += 256) {
      _Float16* buf = (i & 4096) ? kb : qb;
      int r = (i >> 6) & 63;
      int p = i & 63;
      int hd = p >> 4, j = p & 15;
      int c0 = hd * DHEAD + j, c1 = c0 + 16;
      float cv = rct[r * 16 + j], sv = rst[r * 16 + j];
      float a0 = (float)buf[r * EDIM + c0];
      float a1 = (float)buf[r * EDIM + c1];
      buf[r * EDIM + c0] = (_Float16)(a0 * cv - a1 * sv);
      buf[r * EDIM + c1] = (_Float16)(a1 * cv + a0 * sv);
    }
    __syncthreads();

    // ---- attention, per head ----
    for (int hd = 0; hd < NHEAD; ++hd) {
      // scores: 16 tiles, 2 per wave; K = Dh = 32
#pragma unroll
      for (int tc = 0; tc < 2; ++tc) {
        int ti = wid + tc * 8;
        int st = ti >> 2, tt = ti & 3;
        v16h a  = load_a(qb, EDIM, st * 16, hd * DHEAD);
        v16h bf = load_b_kT(kb, tt * 16, hd * DHEAD);
        v8f c;
#pragma unroll
        for (int j = 0; j < 8; ++j) c[j] = 0.f;
        c = wmma_f16(a, bf, c);
        int t    = tt * 16 + (lane & 15);
        float ng = (t >= dC) ? -1e9f : 0.f;
        int m0   = st * 16 + 8 * (lane >> 4);
#pragma unroll
        for (int v = 0; v < 8; ++v) sf[(m0 + v) * SEQ + t] = c[v] * scale + ng;
      }
      __syncthreads();
      // ---- parallel softmax: 4 lanes per row, 16 elems each ----
      {
        int row = tid >> 2, sub = tid & 3;
        const float4* s4 = (const float4*)(sf + row * SEQ + sub * 16);
        float4 w0 = s4[0], w1 = s4[1], w2 = s4[2], w3 = s4[3];
        float mx = fmaxf(fmaxf(fmaxf(w0.x, w0.y), fmaxf(w0.z, w0.w)),
                         fmaxf(fmaxf(w1.x, w1.y), fmaxf(w1.z, w1.w)));
        mx = fmaxf(mx, fmaxf(fmaxf(fmaxf(w2.x, w2.y), fmaxf(w2.z, w2.w)),
                             fmaxf(fmaxf(w3.x, w3.y), fmaxf(w3.z, w3.w))));
        mx = fmaxf(mx, __shfl_xor(mx, 1));
        mx = fmaxf(mx, __shfl_xor(mx, 2));
        float e[16];
        e[0]  = __expf(w0.x - mx); e[1]  = __expf(w0.y - mx);
        e[2]  = __expf(w0.z - mx); e[3]  = __expf(w0.w - mx);
        e[4]  = __expf(w1.x - mx); e[5]  = __expf(w1.y - mx);
        e[6]  = __expf(w1.z - mx); e[7]  = __expf(w1.w - mx);
        e[8]  = __expf(w2.x - mx); e[9]  = __expf(w2.y - mx);
        e[10] = __expf(w2.z - mx); e[11] = __expf(w2.w - mx);
        e[12] = __expf(w3.x - mx); e[13] = __expf(w3.y - mx);
        e[14] = __expf(w3.z - mx); e[15] = __expf(w3.w - mx);
        float sum = 0.f;
#pragma unroll
        for (int i = 0; i < 16; ++i) sum += e[i];
        sum += __shfl_xor(sum, 1);
        sum += __shfl_xor(sum, 2);
        float inv = 1.0f / sum;
        uint4* o4 = (uint4*)(pb + row * SEQ + sub * 16);
        uint4 q0, q1;
        H2U p;
        p.h[0] = (_Float16)(e[0] * inv);  p.h[1] = (_Float16)(e[1] * inv);  q0.x = p.u;
        p.h[0] = (_Float16)(e[2] * inv);  p.h[1] = (_Float16)(e[3] * inv);  q0.y = p.u;
        p.h[0] = (_Float16)(e[4] * inv);  p.h[1] = (_Float16)(e[5] * inv);  q0.z = p.u;
        p.h[0] = (_Float16)(e[6] * inv);  p.h[1] = (_Float16)(e[7] * inv);  q0.w = p.u;
        p.h[0] = (_Float16)(e[8] * inv);  p.h[1] = (_Float16)(e[9] * inv);  q1.x = p.u;
        p.h[0] = (_Float16)(e[10] * inv); p.h[1] = (_Float16)(e[11] * inv); q1.y = p.u;
        p.h[0] = (_Float16)(e[12] * inv); p.h[1] = (_Float16)(e[13] * inv); q1.z = p.u;
        p.h[0] = (_Float16)(e[14] * inv); p.h[1] = (_Float16)(e[15] * inv); q1.w = p.u;
        o4[0] = q0; o4[1] = q1;
      }
      __syncthreads();
      // o = P @ V : one 16x16 tile per wave; K = 64 -> 2 WMMA
      {
        int rt = wid >> 1, ct = wid & 1;
        int nb = hd * DHEAD + ct * 16;
        v8f c;
#pragma unroll
        for (int j = 0; j < 8; ++j) c[j] = 0.f;
#pragma unroll
        for (int kc = 0; kc < 2; ++kc) {
          v16h a  = load_a(pb, SEQ, rt * 16, kc * 32);
          v16h bf = load_b_vT(vt, kc * 32, nb);
          c = wmma_f16(a, bf, c);
        }
        store_h(c, hf, EDIM, rt * 16, nb);
      }
    }
    __syncthreads();

    // ---- x += o @ Wo + bo ----
    {
      int nb = wid * 16;
      v8f acc[4];
      gemm128(hf, EDIM, po, wid, 4, 0, bo[l * EDIM + nb + (lane & 15)], acc);
#pragma unroll
      for (int rt = 0; rt < 4; ++rt) add_xs(acc[rt], xs, rt * 16, nb);
    }
    __syncthreads();

    // ---- LN2 ----
    layer_norm_par(xs, ln2g + l * EDIM, ln2b + l * EDIM, hf, tid);
    __syncthreads();

    // ---- FFN (two 128-wide halves of FF) ----
    for (int hF = 0; hF < 2; ++hF) {
      {
        int nb = wid * 16;
        int gc = hF * 128 + nb + (lane & 15);
        v8f acc[4];
        gemm128(hf, EDIM, p1, hF * 8 + wid, 4, 0, b1[l * FFDIM + gc], acc);
#pragma unroll
        for (int rt = 0; rt < 4; ++rt) {
          v8f c = acc[rt];
#pragma unroll
          for (int v = 0; v < 8; ++v) {
            float x = c[v];
            c[v] = 0.5f * x *
                   (1.0f + tanhf(0.7978845608028654f * (x + 0.044715f * x * x * x)));
          }
          store_h(c, qb, EDIM, rt * 16, nb);  // qb reused as gelu buffer
        }
      }
      __syncthreads();
      {
        int nb = wid * 16;
        float bs = (hF == 0) ? b2[l * EDIM + nb + (lane & 15)] : 0.f;
        v8f acc[4];
        gemm128(qb, EDIM, p2, wid, 8, hF * 4, bs, acc);
#pragma unroll
        for (int rt = 0; rt < 4; ++rt) add_xs(acc[rt], xs, rt * 16, nb);
      }
      __syncthreads();
    }
  }

  // ---- final LN on cls rows, write output ----
  if (tid < CTOK) {
    const float* row = xs + tid * EDIM;
    float s = 0.f, ss = 0.f;
    for (int e = 0; e < EDIM; ++e) { float v = row[e]; s += v; ss += v * v; }
    float m   = s * (1.0f / EDIM);
    float var = ss * (1.0f / EDIM) - m * m;
    float r   = rsqrtf(var + 1e-5f);
    float* orow = out + (size_t)n * (CTOK * EDIM) + tid * EDIM;
    for (int e = 0; e < EDIM; ++e)
      orow[e] = (row[e] - m) * r * olng[e] + olnb[e];
  }
}

extern "C" void kernel_launch(void* const* d_in, const int* in_sizes, int n_in,
                              void* d_out, int out_size, void* d_ws, size_t ws_size,
                              hipStream_t stream) {
  const float* emb  = (const float*)d_in[0];
  const float* cls  = (const float*)d_in[1];
  const float* ln1g = (const float*)d_in[2];
  const float* ln1b = (const float*)d_in[3];
  const float* wq   = (const float*)d_in[4];
  const float* bq   = (const float*)d_in[5];
  const float* wk   = (const float*)d_in[6];
  const float* bk   = (const float*)d_in[7];
  const float* wv   = (const float*)d_in[8];
  const float* bv   = (const float*)d_in[9];
  const float* wo   = (const float*)d_in[10];
  const float* bo   = (const float*)d_in[11];
  const float* ln2g = (const float*)d_in[12];
  const float* ln2b = (const float*)d_in[13];
  const float* w1   = (const float*)d_in[14];
  const float* b1   = (const float*)d_in[15];
  const float* w2   = (const float*)d_in[16];
  const float* b2   = (const float*)d_in[17];
  const float* olng = (const float*)d_in[18];
  const float* olnb = (const float*)d_in[19];
  const int*   dvec = (const int*)d_in[20];

  unsigned* ws = (unsigned*)d_ws;

  auto pp = [&](const float* W, unsigned* dst, int K, int N) {
    int total = NLAYER * (N / 16) * (K / 32) * 256;
    prepack_kernel<<<total / 256, 256, 0, stream>>>(W, dst, K, N);
  };
  pp(wq, ws + OFF_Q,  EDIM,  EDIM);
  pp(wk, ws + OFF_K,  EDIM,  EDIM);
  pp(wv, ws + OFF_V,  EDIM,  EDIM);
  pp(wo, ws + OFF_O,  EDIM,  EDIM);
  pp(w1, ws + OFF_W1, EDIM,  FFDIM);
  pp(w2, ws + OFF_W2, FFDIM, EDIM);

  int NB = in_sizes[0] / (SEQ * EDIM);  // B*T sequences
  int Bb = in_sizes[20];                // batch B
  int TT = NB / Bb;

  vit_cls_kernel<<<NB, 256, 0, stream>>>(emb, cls, ln1g, ln1b, ws, bq, bk, bv,
                                         bo, ln2g, ln2b, b1, b2, olng, olnb,
                                         dvec, (float*)d_out, TT);
}
